// ScalableHyperGNN_72507637891196
// MI455X (gfx1250) — matrix operand order repaired and checked by reading, not compiled, not run on hardware
//
#include <hip/hip_runtime.h>
#include <hip/hip_bf16.h>

// ---------------------------------------------------------------------------
// ScalableHyperGNN forward for MI455X (gfx1250, wave32, WMMA)
//   h = relu over 2 hypergraph layers with hypernetwork modulation
// Dominant costs: 9.8 GFLOP bf16-WMMA projection GEMM + ~1.8 GB of
// gather/scatter traffic (segment means) handled with f32 atomics in L2.
// ---------------------------------------------------------------------------

#define N_NODES 100000
#define N_EDGES 20000
#define N_INC   800000
#define TDIM    384
#define HID     128

typedef __attribute__((ext_vector_type(16))) __bf16         v16bf;
typedef __attribute__((ext_vector_type(8)))  float          v8f;
typedef __attribute__((ext_vector_type(16))) unsigned short v16u;
typedef __attribute__((ext_vector_type(8)))  unsigned short v8u;

__device__ __forceinline__ unsigned short f2bf(float f) {
    // round-to-nearest-even f32 -> bf16
    unsigned int u = __float_as_uint(f);
    u += 0x7FFFu + ((u >> 16) & 1u);
    return (unsigned short)(u >> 16);
}

// ---------------------------------------------------------------------------
// Projection GEMM: H[N,128] = X[N,384] @ W[384,128] + bias
// Block = 256 threads = 8 wave32. Each wave computes a 16-row x 128-col strip.
// K loop: 12 chunks of 32. A/B staged in LDS as bf16 (RNE).
// WMMA: v_wmma_f32_16x16x32_bf16, f32 accumulate.
// ---------------------------------------------------------------------------
__global__ __launch_bounds__(256)
void proj_gemm(const float* __restrict__ X, const float* __restrict__ W,
               const float* __restrict__ bias, float* __restrict__ H,
               int n_rows) {
    // row stride 40 ushorts = 80 bytes -> keeps 16B alignment for b128 LDS loads
    __shared__ __align__(16) unsigned short As[128][40];
    __shared__ __align__(16) unsigned short Bs[128][40];

    const int tid  = threadIdx.x;
    const int lane = tid & 31;
    const int wv   = tid >> 5;
    const int row0 = blockIdx.x * 128;

    v8f acc[8];
#pragma unroll
    for (int i = 0; i < 8; ++i) {
        v8f z = {0.f, 0.f, 0.f, 0.f, 0.f, 0.f, 0.f, 0.f};
        acc[i] = z;
    }

    // ISA 7.12.2 16-bit A 16x32 layout:
    //   lanes 0-15 : row M=lane,    elems 0..7 = K0..7,  elems 8..15 = K16..23
    //   lanes 16-31: row M=lane-16, elems 0..7 = K8..15, elems 8..15 = K24..31
    const int arow  = wv * 16 + (lane & 15);
    const int akoff = (lane & 16) ? 8 : 0;
    // 16-bit B 32x16 layout: lanes 0-15 col N=lane K0..7,K16..23; lanes 16-31 K8..15,K24..31
    const int bkoff = (lane & 16) ? 8 : 0;
    const int c15   = lane & 15;

    for (int kb = 0; kb < TDIM; kb += 32) {
        __syncthreads();
        // --- stage A tile: 128 rows x 32 k, f32 -> bf16 ---
#pragma unroll
        for (int j = 0; j < 4; ++j) {
            int idx  = tid + 256 * j;   // 1024 float4's
            int r    = idx >> 3;
            int c4   = idx & 7;
            int grow = row0 + r;
            float4 v = make_float4(0.f, 0.f, 0.f, 0.f);
            if (grow < n_rows)
                v = *(const float4*)&X[(size_t)grow * TDIM + kb + c4 * 4];
            unsigned int lo = (unsigned)f2bf(v.x) | ((unsigned)f2bf(v.y) << 16);
            unsigned int hi = (unsigned)f2bf(v.z) | ((unsigned)f2bf(v.w) << 16);
            *(uint2*)&As[r][c4 * 4] = make_uint2(lo, hi);
        }
        // --- stage B tile transposed: Bs[n][k] = bf16(W[kb+k][n]) ---
#pragma unroll
        for (int j = 0; j < 4; ++j) {
            int idx = tid + 256 * j;    // 1024 float4's over 32k x 128n
            int k   = idx >> 5;
            int n4  = idx & 31;
            float4 v = *(const float4*)&W[(size_t)(kb + k) * HID + n4 * 4];
            Bs[n4 * 4 + 0][k] = f2bf(v.x);
            Bs[n4 * 4 + 1][k] = f2bf(v.y);
            Bs[n4 * 4 + 2][k] = f2bf(v.z);
            Bs[n4 * 4 + 3][k] = f2bf(v.w);
        }
        __syncthreads();

        // --- build A fragment ---
        v8u alo = *(const v8u*)&As[arow][akoff];
        v8u ahi = *(const v8u*)&As[arow][akoff + 16];
        v16u afr;
#pragma unroll
        for (int i = 0; i < 8; ++i) { afr[i] = alo[i]; afr[i + 8] = ahi[i]; }

        // --- 8 column tiles, one WMMA each ---
#pragma unroll
        for (int ct = 0; ct < 8; ++ct) {
            int bn  = ct * 16 + c15;
            v8u blo = *(const v8u*)&Bs[bn][bkoff];
            v8u bhi = *(const v8u*)&Bs[bn][bkoff + 16];
            v16u bfr;
#pragma unroll
            for (int i = 0; i < 8; ++i) { bfr[i] = blo[i]; bfr[i + 8] = bhi[i]; }
            acc[ct] = __builtin_amdgcn_wmma_f32_16x16x32_bf16(
                false, __builtin_bit_cast(v16bf, afr),
                false, __builtin_bit_cast(v16bf, bfr),
                (short)0, acc[ct], false, false);
        }
    }

    // --- epilogue: + bias, store (C layout: vgpr r -> row r / r+8) ---
    const int gRow = row0 + wv * 16;
    if (gRow >= n_rows) return;          // wave-uniform guard (N % 16 == 0)
    const int rsub = (lane & 16) ? 8 : 0;
#pragma unroll
    for (int ct = 0; ct < 8; ++ct) {
        int col = ct * 16 + c15;
        float b = bias[col];
#pragma unroll
        for (int r = 0; r < 8; ++r)
            H[(size_t)(gRow + r + rsub) * HID + col] = acc[ct][r] + b;
    }
}

// ---------------------------------------------------------------------------
__global__ void zero_f32(float* __restrict__ p, int n) {
    int i = blockIdx.x * 256 + threadIdx.x;
    if (i < n) p[i] = 0.f;
}

// column sums of h[N,128] -> gsum[128] (atomic partials)
__global__ void colsum(const float* __restrict__ h, float* __restrict__ gsum, int n) {
    int c   = threadIdx.x & 127;
    int sub = threadIdx.x >> 7;  // 0..1 (256 threads cover 2 rows per step)
    float a = 0.f;
    for (int r = blockIdx.x * 2 + sub; r < n; r += gridDim.x * 2)
        a += h[(size_t)r * HID + c];
    atomicAdd(&gsum[c], a);
}

// scale = relu((gsum/N) @ W1 + b1) @ W2 + b2   (one block, 128 threads)
__global__ void scale_gen(const float* __restrict__ gsum,
                          const float* __restrict__ W1, const float* __restrict__ b1,
                          const float* __restrict__ W2, const float* __restrict__ b2,
                          float* __restrict__ scl, float invN) {
    __shared__ float g[HID], t[HID];
    int tid = threadIdx.x;
    g[tid] = gsum[tid] * invN;
    __syncthreads();
    float a = b1[tid];
    for (int k = 0; k < HID; ++k) a += g[k] * W1[k * HID + tid];
    t[tid] = fmaxf(a, 0.f);
    __syncthreads();
    float s = b2[tid];
    for (int k = 0; k < HID; ++k) s += t[k] * W2[k * HID + tid];
    scl[tid] = s;
}

// one wave32 per incidence pair: gather float4 row, atomic-scatter into sum
__global__ __launch_bounds__(256)
void scatter_accum(const float* __restrict__ src,
                   const int* __restrict__ gidx, const int* __restrict__ sidx,
                   float* __restrict__ sum, float* __restrict__ cnt, int n_inc) {
    int inc = blockIdx.x * 8 + (threadIdx.x >> 5);
    if (inc >= n_inc) return;
    int lane = threadIdx.x & 31;
    int g = gidx[inc];
    int s = sidx[inc];
    float4 v = *(const float4*)&src[(size_t)g * HID + lane * 4];
    float* d = &sum[(size_t)s * HID + lane * 4];
    atomicAdd(d + 0, v.x);
    atomicAdd(d + 1, v.y);
    atomicAdd(d + 2, v.z);
    atomicAdd(d + 3, v.w);
    if (lane == 0) atomicAdd(&cnt[s], 1.0f);
}

// sum[i] /= max(cnt[i/128], 1)  in place
__global__ void norm_rows(float* __restrict__ sum, const float* __restrict__ cnt, int n) {
    int i = blockIdx.x * 256 + threadIdx.x;
    if (i >= n) return;
    sum[i] = sum[i] / fmaxf(cnt[i >> 7], 1.0f);
}

// out = relu((msum/max(mcnt,1)) * scale)
__global__ void finalize(const float* __restrict__ msum, const float* __restrict__ mcnt,
                         const float* __restrict__ scl, float* __restrict__ out, int n) {
    int i = blockIdx.x * 256 + threadIdx.x;
    if (i >= n) return;
    float v = msum[i] / fmaxf(mcnt[i >> 7], 1.0f) * scl[i & 127];
    out[i] = fmaxf(v, 0.f);
}

// ---------------------------------------------------------------------------
extern "C" void kernel_launch(void* const* d_in, const int* in_sizes, int n_in,
                              void* d_out, int out_size, void* d_ws, size_t ws_size,
                              hipStream_t stream) {
    const float* text_emb = (const float*)d_in[0];
    const float* W_proj   = (const float*)d_in[1];
    const float* b_proj   = (const float*)d_in[2];
    const float* Wg[2][4] = {
        {(const float*)d_in[3], (const float*)d_in[4], (const float*)d_in[5], (const float*)d_in[6]},
        {(const float*)d_in[7], (const float*)d_in[8], (const float*)d_in[9], (const float*)d_in[10]}};
    const int* node_idx = (const int*)d_in[11];
    const int* edge_idx = (const int*)d_in[12];
    float* out = (float*)d_out;

    const size_t NH = (size_t)N_NODES * HID;  // 12,800,000
    const size_t EH = (size_t)N_EDGES * HID;  //  2,560,000

    float* hbuf  = (float*)d_ws;
    float* m_sum = hbuf + NH;
    float* m_cnt = m_sum + NH;       // contiguous with m_sum for one-shot zero
    float* e_sum = m_cnt + N_NODES;
    float* e_cnt = e_sum + EH;       // contiguous with e_sum
    float* gsum  = e_cnt + N_EDGES;
    float* scl   = gsum + HID;

    // 1) projection GEMM (bf16 WMMA, f32 accumulate)
    {
        int nblocks = (N_NODES + 127) / 128;   // 782
        proj_gemm<<<nblocks, 256, 0, stream>>>(text_emb, W_proj, b_proj, hbuf, N_NODES);
    }

    // 2) two hypergraph layers
    for (int L = 0; L < 2; ++L) {
        // hypernetwork scale
        zero_f32<<<1, 256, 0, stream>>>(gsum, HID);
        colsum<<<400, 256, 0, stream>>>(hbuf, gsum, N_NODES);
        scale_gen<<<1, HID, 0, stream>>>(gsum, Wg[L][0], Wg[L][1], Wg[L][2], Wg[L][3],
                                         scl, 1.0f / (float)N_NODES);

        // node -> edge segment mean
        {
            int ze = (int)(EH + N_EDGES);
            zero_f32<<<(ze + 255) / 256, 256, 0, stream>>>(e_sum, ze);
            scatter_accum<<<(N_INC + 7) / 8, 256, 0, stream>>>(hbuf, node_idx, edge_idx,
                                                               e_sum, e_cnt, N_INC);
            norm_rows<<<((int)EH + 255) / 256, 256, 0, stream>>>(e_sum, e_cnt, (int)EH);
        }

        // edge -> node segment mean, then relu(m * scale)
        {
            int zm = (int)(NH + N_NODES);
            zero_f32<<<(zm + 255) / 256, 256, 0, stream>>>(m_sum, zm);
            scatter_accum<<<(N_INC + 7) / 8, 256, 0, stream>>>(e_sum, edge_idx, node_idx,
                                                               m_sum, m_cnt, N_INC);
            float* dst = (L == 0) ? hbuf : out;
            finalize<<<((int)NH + 255) / 256, 256, 0, stream>>>(m_sum, m_cnt, scl, dst, (int)NH);
        }
    }
}
